// PegSolitaireGNN_36524401885973
// MI455X (gfx1250) — compile-verified, matrix operand order checked by reading, not run on hardware
//
#include <hip/hip_runtime.h>
#include <hip/hip_bf16.h>
#include <math.h>

typedef __bf16 bf16;
typedef __attribute__((ext_vector_type(16))) __bf16 v16bf;
typedef __attribute__((ext_vector_type(8)))  __bf16 v8bf;
typedef __attribute__((ext_vector_type(8)))  float  v8f;

#define NN  50000
#define EE  800000
#define EP  (EE + NN)
#define GG  64
#define RSMAX 256

static inline int cdiv(long a, long b) { return (int)((a + b - 1) / b); }

// ---------------- generic fill ----------------
__global__ void k_fill(float* __restrict__ p, float v, long n) {
    long i = (long)blockIdx.x * blockDim.x + threadIdx.x;
    if (i < n) p[i] = v;
}

// ---------------- layer-1 transform (K=4, scalar) ----------------
__global__ void k_l1(const float* __restrict__ x,
                     const float* __restrict__ Wl, const float* __restrict__ bl,
                     const float* __restrict__ Wr, const float* __restrict__ br,
                     float* __restrict__ xl, float* __restrict__ xr) {
    long i = (long)blockIdx.x * blockDim.x + threadIdx.x;
    if (i >= (long)NN * 256) return;
    int n = (int)(i >> 8), j = (int)(i & 255);
    float x0 = x[n * 4 + 0], x1 = x[n * 4 + 1], x2 = x[n * 4 + 2], x3 = x[n * 4 + 3];
    xl[i] = x0 * Wl[j] + x1 * Wl[256 + j] + x2 * Wl[512 + j] + x3 * Wl[768 + j] + bl[j];
    xr[i] = x0 * Wr[j] + x1 * Wr[256 + j] + x2 * Wr[512 + j] + x3 * Wr[768 + j] + br[j];
}

// ---------------- weight transpose + f32->bf16 ----------------
// W: [K, Ncols] row-major f32  ->  Wt: [Ncols, K] row-major bf16
__global__ void k_wt(const float* __restrict__ W, bf16* __restrict__ Wt, int K, int Ncols) {
    long i = (long)blockIdx.x * blockDim.x + threadIdx.x;
    if (i >= (long)K * Ncols) return;
    int k = (int)(i / Ncols), j = (int)(i % Ncols);
    Wt[(long)j * K + k] = (bf16)W[i];
}

// ---------------- float atomic max (CAS) ----------------
__device__ __forceinline__ void atomicMaxF(float* addr, float v) {
    int old = __float_as_int(*addr);
    while (__int_as_float(old) < v) {
        int assumed = old;
        old = atomicCAS((int*)addr, assumed, __float_as_int(v));
        if (old == assumed) break;
    }
}

// ---------------- edge pass 1: logits + segment max (wave per edge) ----------------
__global__ void k_edge_logits(const int* __restrict__ ei,
                              const float* __restrict__ xl, const float* __restrict__ xr,
                              const float* __restrict__ att,
                              float* __restrict__ logits, float* __restrict__ segmax,
                              int H, int C) {
    int wid  = (int)(((long)blockIdx.x * blockDim.x + threadIdx.x) >> 5);
    int lane = threadIdx.x & 31;
    if (wid >= EP) return;
    int s, d;
    if (wid < EE) { s = ei[wid]; d = ei[EE + wid]; } else { s = d = wid - EE; }
    int RS  = H * C;
    int cpl = C >> 5; // channels per lane (2 for C=64)
    for (int h = 0; h < H; ++h) {
        float t = 0.f;
        long bl = (long)s * RS + h * C;
        long br = (long)d * RS + h * C;
        int  ab = h * C;
        for (int q = 0; q < cpl; ++q) {
            int c = lane * cpl + q;
            float v = xl[bl + c] + xr[br + c];
            v = v > 0.f ? v : 0.2f * v;
            t += att[ab + c] * v;
        }
        for (int o = 16; o; o >>= 1) t += __shfl_xor(t, o, 32);
        if (lane == 0) {
            logits[(long)wid * H + h] = t;
            atomicMaxF(&segmax[(long)d * H + h], t);
        }
    }
}

// ---------------- edge pass 2: exp + segment sum (thread per edge*head) ----------------
__global__ void k_edge_expsum(const int* __restrict__ ei,
                              float* __restrict__ logits,
                              const float* __restrict__ segmax, float* __restrict__ segsum,
                              int H) {
    long i = (long)blockIdx.x * blockDim.x + threadIdx.x;
    if (i >= (long)EP * H) return;
    int e = (int)(i / H), h = (int)(i % H);
    int d = (e < EE) ? ei[EE + e] : e - EE;
    float a = __expf(logits[i] - segmax[(long)d * H + h]);
    logits[i] = a;
    atomicAdd(&segsum[(long)d * H + h], a);
}

// ---------------- edge pass 3: scatter alpha * xl[src] (wave per edge) ----------------
__global__ void k_edge_scatter(const int* __restrict__ ei,
                               const float* __restrict__ logits, const float* __restrict__ segsum,
                               const float* __restrict__ xl, float* __restrict__ acc,
                               int H, int C) {
    int wid  = (int)(((long)blockIdx.x * blockDim.x + threadIdx.x) >> 5);
    int lane = threadIdx.x & 31;
    if (wid >= EP) return;
    int s, d;
    if (wid < EE) { s = ei[wid]; d = ei[EE + wid]; } else { s = d = wid - EE; }
    int RS  = H * C;
    int cpl = C >> 5;
    for (int h = 0; h < H; ++h) {
        float alpha = logits[(long)wid * H + h] / (segsum[(long)d * H + h] + 1e-16f);
        long bs = (long)s * RS + h * C;
        long bd = (long)d * RS + h * C;
        for (int q = 0; q < cpl; ++q) {
            int c = lane * cpl + q;
            atomicAdd(&acc[bd + c], alpha * xl[bs + c]);
        }
    }
}

// ---------------- bias + ELU, optional bf16 / f32 output ----------------
__global__ void k_bias_elu(const float* __restrict__ acc, const float* __restrict__ bias,
                           int RS, bf16* __restrict__ out_bf, float* __restrict__ out_f) {
    long i = (long)blockIdx.x * blockDim.x + threadIdx.x;
    if (i >= (long)NN * RS) return;
    float v = acc[i] + bias[i % RS];
    float e = v > 0.f ? v : (__expf(v) - 1.f);
    if (out_bf) out_bf[i] = (bf16)e;
    if (out_f)  out_f[i]  = e;
}

// ---------------- WMMA bf16 GEMM: out[N,Ncols] = A[N,K] * Bt[Ncols,K]^T + bias ----------------
// One wave computes a 16x16 output tile via v_wmma_f32_16x16x32_bf16.
__global__ void k_gemm(const bf16* __restrict__ A, const bf16* __restrict__ Bt,
                       const float* __restrict__ bias, float* __restrict__ out,
                       int Nrows, int K, int Ncols) {
    int wid  = (int)(((long)blockIdx.x * blockDim.x + threadIdx.x) >> 5);
    int lane = threadIdx.x & 31;
    int mtiles = Nrows >> 4, ntiles = Ncols >> 4;
    if (wid >= mtiles * ntiles) return;
    int mt = wid / ntiles, nt = wid % ntiles;
    int m0 = mt << 4, n0 = nt << 4;
    int ln = lane & 15;
    int kb = (lane >> 4) << 3;              // 0 or 8 (ISA 16-bit A/B K-group split)
    const bf16* arow = A  + (long)(m0 + ln) * K + kb;
    const bf16* brow = Bt + (long)(n0 + ln) * K + kb;
    v8f c = {};
#pragma unroll
    for (int k0 = 0; k0 < 256; k0 += 32) {
        v8bf a0 = *(const v8bf*)(arow + k0);
        v8bf a1 = *(const v8bf*)(arow + k0 + 16);
        v8bf b0 = *(const v8bf*)(brow + k0);
        v8bf b1 = *(const v8bf*)(brow + k0 + 16);
        v16bf a, b;
#pragma unroll
        for (int q = 0; q < 8; ++q) { a[q] = a0[q]; a[q + 8] = a1[q]; b[q] = b0[q]; b[q + 8] = b1[q]; }
        c = __builtin_amdgcn_wmma_f32_16x16x32_bf16(false, a, false, b, (short)0, c, false, false);
    }
    int mo = (lane >> 4) << 3;              // C/D layout: lanes 16-31 hold M+8
    float bcol = bias[n0 + ln];
#pragma unroll
    for (int r = 0; r < 8; ++r)
        out[(long)(m0 + r + mo) * Ncols + n0 + ln] = c[r] + bcol;
}

// ---------------- global mean pool (sum + count) ----------------
__global__ void k_pool(const float* __restrict__ h3, const int* __restrict__ batch,
                       float* __restrict__ ssum, float* __restrict__ cnt) {
    long i = (long)blockIdx.x * blockDim.x + threadIdx.x;
    if (i >= (long)NN * 64) return;
    int n = (int)(i >> 6), c = (int)(i & 63);
    int g = batch[n];
    atomicAdd(&ssum[g * 64 + c], h3[i]);
    if (c == 0) atomicAdd(&cnt[g], 1.f);
}

// ---------------- MLP heads, one block per graph ----------------
__global__ void k_heads(const float* __restrict__ ssum, const float* __restrict__ cnt,
                        const float* __restrict__ pegs, const float* __restrict__ movec,
                        const float* __restrict__ vW1, const float* __restrict__ vb1,
                        const float* __restrict__ vW2, const float* __restrict__ vb2,
                        const float* __restrict__ pW1, const float* __restrict__ pb1,
                        const float* __restrict__ pW2, const float* __restrict__ pb2,
                        float* __restrict__ out) {
    int g = blockIdx.x, t = threadIdx.x;
    __shared__ float comb[66], hv[64], hp[64];
    if (t < 64) comb[t] = ssum[g * 64 + t] / fmaxf(cnt[g], 1.f);
    if (t == 64) comb[64] = pegs[g];
    if (t == 65) comb[65] = movec[g];
    __syncthreads();
    if (t < 64) {
        float sv = vb1[t], sp = pb1[t];
        for (int i = 0; i < 66; ++i) { float ci = comb[i]; sv += ci * vW1[i * 64 + t]; sp += ci * pW1[i * 64 + t]; }
        hv[t] = fmaxf(sv, 0.f); hp[t] = fmaxf(sp, 0.f);
    }
    __syncthreads();
    if (t < 196) {
        float s = pb2[t];
        for (int j = 0; j < 64; ++j) s += hp[j] * pW2[j * 196 + t];
        out[g * 196 + t] = s;
    }
    if (t == 0) {
        float s = vb2[0];
        for (int j = 0; j < 64; ++j) s += hv[j] * vW2[j];
        out[GG * 196 + g] = tanhf(s);
    }
}

// ---------------- workspace layout (bytes) ----------------
#define OFF_XL     0L
#define OFF_XR     51200000L            // acc aliases this (xr dead after pass1)
#define OFF_HBF    102400000L
#define OFF_LOG    128000000L
#define OFF_SMAX   141600000L
#define OFF_SSUM   142400000L
#define OFF_WL2T   143200000L
#define OFF_WR2T   143331072L
#define OFF_WL3T   143462144L
#define OFF_WR3T   143494912L
#define OFF_PSUM   143527680L
#define OFF_CNT    143544064L

extern "C" void kernel_launch(void* const* d_in, const int* in_sizes, int n_in,
                              void* d_out, int out_size, void* d_ws, size_t ws_size,
                              hipStream_t stream) {
    (void)in_sizes; (void)n_in; (void)out_size; (void)ws_size;
    const float* x     = (const float*)d_in[0];
    const int*   ei    = (const int*)d_in[1];
    const int*   batch = (const int*)d_in[2];
    const float* pegs  = (const float*)d_in[3];
    const float* movec = (const float*)d_in[4];
    const float *Wl1=(const float*)d_in[5],  *bl1=(const float*)d_in[6],
                *Wr1=(const float*)d_in[7],  *br1=(const float*)d_in[8],
                *att1=(const float*)d_in[9], *bias1=(const float*)d_in[10],
                *Wl2=(const float*)d_in[11], *bl2=(const float*)d_in[12],
                *Wr2=(const float*)d_in[13], *br2=(const float*)d_in[14],
                *att2=(const float*)d_in[15],*bias2=(const float*)d_in[16],
                *Wl3=(const float*)d_in[17], *bl3=(const float*)d_in[18],
                *Wr3=(const float*)d_in[19], *br3=(const float*)d_in[20],
                *att3=(const float*)d_in[21],*bias3=(const float*)d_in[22],
                *vW1=(const float*)d_in[23], *vb1=(const float*)d_in[24],
                *vW2=(const float*)d_in[25], *vb2=(const float*)d_in[26],
                *pW1=(const float*)d_in[27], *pb1=(const float*)d_in[28],
                *pW2=(const float*)d_in[29], *pb2=(const float*)d_in[30];
    char* ws = (char*)d_ws;
    float* xl     = (float*)(ws + OFF_XL);
    float* xr     = (float*)(ws + OFF_XR);
    float* acc    = xr;                       // alias: xr dead once logits computed
    bf16*  hbf    = (bf16*)(ws + OFF_HBF);
    float* logits = (float*)(ws + OFF_LOG);
    float* smax   = (float*)(ws + OFF_SMAX);
    float* ssum   = (float*)(ws + OFF_SSUM);
    bf16*  wl2t   = (bf16*)(ws + OFF_WL2T);
    bf16*  wr2t   = (bf16*)(ws + OFF_WR2T);
    bf16*  wl3t   = (bf16*)(ws + OFF_WL3T);
    bf16*  wr3t   = (bf16*)(ws + OFF_WR3T);
    float* psum   = (float*)(ws + OFF_PSUM);
    float* cnt    = (float*)(ws + OFF_CNT);
    float* out    = (float*)d_out;
    float* h3     = xl;                       // layer-3 activations reuse xl

    const int B = 256;
    const int edgeWaveBlocks = cdiv((long)EP * 32, B);

    // weight prep
    k_wt<<<cdiv(256L*256, B), B, 0, stream>>>(Wl2, wl2t, 256, 256);
    k_wt<<<cdiv(256L*256, B), B, 0, stream>>>(Wr2, wr2t, 256, 256);
    k_wt<<<cdiv(256L*64,  B), B, 0, stream>>>(Wl3, wl3t, 256, 64);
    k_wt<<<cdiv(256L*64,  B), B, 0, stream>>>(Wr3, wr3t, 256, 64);

    // ---- layer 1 ----
    k_l1<<<cdiv((long)NN*256, B), B, 0, stream>>>(x, Wl1, bl1, Wr1, br1, xl, xr);
    k_fill<<<cdiv((long)NN*4, B), B, 0, stream>>>(smax, -3.0e38f, (long)NN*4);
    k_fill<<<cdiv((long)NN*4, B), B, 0, stream>>>(ssum, 0.f, (long)NN*4);
    k_edge_logits<<<edgeWaveBlocks, B, 0, stream>>>(ei, xl, xr, att1, logits, smax, 4, 64);
    k_edge_expsum<<<cdiv((long)EP*4, B), B, 0, stream>>>(ei, logits, smax, ssum, 4);
    k_fill<<<cdiv((long)NN*256, B), B, 0, stream>>>(acc, 0.f, (long)NN*256);
    k_edge_scatter<<<edgeWaveBlocks, B, 0, stream>>>(ei, logits, ssum, xl, acc, 4, 64);
    k_bias_elu<<<cdiv((long)NN*256, B), B, 0, stream>>>(acc, bias1, 256, hbf, nullptr);

    // ---- layer 2 ----
    k_gemm<<<cdiv((long)(NN/16)*(256/16)*32, B), B, 0, stream>>>(hbf, wl2t, bl2, xl, NN, 256, 256);
    k_gemm<<<cdiv((long)(NN/16)*(256/16)*32, B), B, 0, stream>>>(hbf, wr2t, br2, xr, NN, 256, 256);
    k_fill<<<cdiv((long)NN*4, B), B, 0, stream>>>(smax, -3.0e38f, (long)NN*4);
    k_fill<<<cdiv((long)NN*4, B), B, 0, stream>>>(ssum, 0.f, (long)NN*4);
    k_edge_logits<<<edgeWaveBlocks, B, 0, stream>>>(ei, xl, xr, att2, logits, smax, 4, 64);
    k_edge_expsum<<<cdiv((long)EP*4, B), B, 0, stream>>>(ei, logits, smax, ssum, 4);
    k_fill<<<cdiv((long)NN*256, B), B, 0, stream>>>(acc, 0.f, (long)NN*256);
    k_edge_scatter<<<edgeWaveBlocks, B, 0, stream>>>(ei, logits, ssum, xl, acc, 4, 64);
    k_bias_elu<<<cdiv((long)NN*256, B), B, 0, stream>>>(acc, bias2, 256, hbf, nullptr);

    // ---- layer 3 (H=1, C=64) ----
    k_gemm<<<cdiv((long)(NN/16)*(64/16)*32, B), B, 0, stream>>>(hbf, wl3t, bl3, xl, NN, 256, 64);
    k_gemm<<<cdiv((long)(NN/16)*(64/16)*32, B), B, 0, stream>>>(hbf, wr3t, br3, xr, NN, 256, 64);
    k_fill<<<cdiv((long)NN, B), B, 0, stream>>>(smax, -3.0e38f, (long)NN);
    k_fill<<<cdiv((long)NN, B), B, 0, stream>>>(ssum, 0.f, (long)NN);
    k_edge_logits<<<edgeWaveBlocks, B, 0, stream>>>(ei, xl, xr, att3, logits, smax, 1, 64);
    k_edge_expsum<<<cdiv((long)EP, B), B, 0, stream>>>(ei, logits, smax, ssum, 1);
    k_fill<<<cdiv((long)NN*64, B), B, 0, stream>>>(acc, 0.f, (long)NN*64);
    k_edge_scatter<<<edgeWaveBlocks, B, 0, stream>>>(ei, logits, ssum, xl, acc, 1, 64);
    k_bias_elu<<<cdiv((long)NN*64, B), B, 0, stream>>>(acc, bias3, 64, nullptr, h3);

    // ---- pool + heads ----
    k_fill<<<cdiv((long)GG*64 + GG, B), B, 0, stream>>>(psum, 0.f, (long)GG*64 + GG); // psum then cnt contiguous
    k_pool<<<cdiv((long)NN*64, B), B, 0, stream>>>(h3, batch, psum, cnt);
    k_heads<<<GG, 256, 0, stream>>>(psum, cnt, pegs, movec,
                                    vW1, vb1, vW2, vb2, pW1, pb1, pW2, pb2, out);
}
